// SparseConvNet_29180007809610
// MI455X (gfx1250) — matrix operand order repaired and checked by reading, not compiled, not run on hardware
//
#include <hip/hip_runtime.h>
#include <hip/hip_bf16.h>

// ---------------------------------------------------------------------------
// Types for CDNA5 WMMA (wave32): D(16x16 f32) = A(16x32 bf16) * B(32x16 bf16) + C
// ---------------------------------------------------------------------------
typedef __attribute__((ext_vector_type(16))) __bf16 v16bf;
typedef __attribute__((ext_vector_type(8)))  float  v8f;

union Frag16 {
    v16bf bf;
    unsigned int u[8];
};

__device__ __forceinline__ unsigned short f2bf(float f) {
    unsigned int u = __float_as_uint(f);
    unsigned int r = (u + 0x7FFFu + ((u >> 16) & 1u)) >> 16;   // RNE
    return (unsigned short)r;
}
__device__ __forceinline__ float bf2f_lo(unsigned int d) {   // low 16 bits
    return __uint_as_float(d << 16);
}
__device__ __forceinline__ float bf2f_hi(unsigned int d) {   // high 16 bits
    return __uint_as_float(d & 0xFFFF0000u);
}

// ---------------------------------------------------------------------------
// Pack x * mask into channels-last bf16:  buf[v*32 + c] = bf16(x[c*V + v] * m[v])
// ---------------------------------------------------------------------------
__global__ void x_pack(const float* __restrict__ x, const int* __restrict__ mask,
                       unsigned short* __restrict__ buf, int V) {
    int i = blockIdx.x * blockDim.x + threadIdx.x;
    if (i >= 32 * V) return;
    int c = i / V;
    int v = i - c * V;
    float m = mask[v] ? 1.0f : 0.0f;
    buf[v * 32 + c] = f2bf(x[i] * m);
}

__global__ void m_pack(const int* __restrict__ mask, float* __restrict__ m0, int V) {
    int v = blockIdx.x * blockDim.x + threadIdx.x;
    if (v < V) m0[v] = mask[v] ? 1.0f : 0.0f;
}

// 3x3x3 stride-2 pad-1 max-pool > 0
__global__ void m_down(const float* __restrict__ m0, float* __restrict__ m1,
                       int Din, int Dout) {
    int v = blockIdx.x * blockDim.x + threadIdx.x;
    int Vo = Dout * Dout * Dout;
    if (v >= Vo) return;
    int x = v % Dout;
    int t = v / Dout;
    int y = t % Dout;
    int z = t / Dout;
    float mx = 0.0f;
    for (int dz = 0; dz < 3; ++dz) {
        int iz = 2 * z + dz - 1;
        if (iz < 0 || iz >= Din) continue;
        for (int dy = 0; dy < 3; ++dy) {
            int iy = 2 * y + dy - 1;
            if (iy < 0 || iy >= Din) continue;
            for (int dx = 0; dx < 3; ++dx) {
                int ix = 2 * x + dx - 1;
                if (ix < 0 || ix >= Din) continue;
                mx = fmaxf(mx, m0[(iz * Din + iy) * Din + ix]);
            }
        }
    }
    m1[v] = (mx > 0.0f) ? 1.0f : 0.0f;
}

// w (5,co,ci,3,3,3) f32 -> wbf[l][tap][co][ci] bf16
__global__ void w_pack(const float* __restrict__ w, unsigned short* __restrict__ wbf) {
    const int total = 5 * 32 * 32 * 27;
    int i = blockIdx.x * blockDim.x + threadIdx.x;
    if (i >= total) return;
    int tap = i % 27;
    int t   = i / 27;
    int ci  = t % 32;  t /= 32;
    int co  = t % 32;
    int l   = t / 32;
    wbf[((l * 27 + tap) * 32 + co) * 32 + ci] = f2bf(w[i]);
}

// fold BN:  sb[l*64 + c] = gamma/sqrt(var+eps);  sb[l*64+32+c] = beta - mean*scale
__global__ void bn_pack(const float* __restrict__ g,  const float* __restrict__ be,
                        const float* __restrict__ me, const float* __restrict__ va,
                        float* __restrict__ sb) {
    int i = blockIdx.x * blockDim.x + threadIdx.x;
    if (i >= 5 * 32) return;
    float s = g[i] * rsqrtf(va[i] + 0.001f);
    int l = i / 32, c = i % 32;
    sb[l * 64 + c]      = s;
    sb[l * 64 + 32 + c] = be[i] - me[i] * s;
}

// ---------------------------------------------------------------------------
// Implicit-GEMM conv3d 3x3x3 (32->32 ch) + BN + ReLU + mask, via WMMA bf16.
// Input/output channels-last bf16 (row = 32 ch = 64B). One wave computes a
// 16-voxel x 32-cout tile: 27 taps x 2 x v_wmma_f32_16x16x32_bf16.
// ---------------------------------------------------------------------------
__global__ void __launch_bounds__(256)
conv3_wmma(const unsigned short* __restrict__ in,
           unsigned short* __restrict__ out,
           const unsigned short* __restrict__ wly,   // [27][32][32] bf16
           const float* __restrict__ sb,             // [2][32] scale,bias
           const float* __restrict__ mko,            // mask at output res
           int Din, int Dout, int stride) {
    __shared__ unsigned int sw[27 * 32 * 16];        // 55296 B of bf16 weights

    // cooperative stage of this layer's weights into LDS
    const unsigned int* wsrc = (const unsigned int*)wly;
    for (int i = threadIdx.x; i < 27 * 32 * 16; i += 256) sw[i] = wsrc[i];
    __syncthreads();

    const int lane = threadIdx.x & 31;
    const int wave = threadIdx.x >> 5;
    const int half = lane >> 4;          // 0: lanes 0-15, 1: lanes 16-31
    const int mrow = lane & 15;
    const int voxBase = (blockIdx.x * 8 + wave) * 16;

    // A-matrix: this lane supplies row M = mrow (its voxel)
    int vA = voxBase + mrow;
    int ax = vA % Dout;
    int tt = vA / Dout;
    int ay = tt % Dout;
    int az = tt / Dout;
    int bx0 = ax * stride - 1;
    int by0 = ay * stride - 1;
    int bz0 = az * stride - 1;

    const int Kb = mrow + (half << 4);   // B-fragment K index for this lane
    const int dwBase = half * 4;         // A-fragment dword offset (K +8 for upper half)
    const unsigned int* inw = (const unsigned int*)in;

    v8f acc0 = {};   // cout 0..15
    v8f acc1 = {};   // cout 16..31

    for (int dz = 0; dz < 3; ++dz) {
        int iz = bz0 + dz;
        bool okz = (iz >= 0) && (iz < Din);
        for (int dy = 0; dy < 3; ++dy) {
            int iy = by0 + dy;
            bool oky = okz && (iy >= 0) && (iy < Din);
#pragma unroll
            for (int dx = 0; dx < 3; ++dx) {
                int ix = bx0 + dx;
                bool ok = oky && (ix >= 0) && (ix < Din);

                Frag16 a;
                if (ok) {
                    const unsigned int* p = inw + ((size_t)((iz * Din + iy) * Din + ix)) * 16;
                    __builtin_prefetch((const void*)(p + 16), 0, 1);  // next row
#pragma unroll
                    for (int v = 0; v < 8; ++v)
                        a.u[v] = p[v + (v & 4) + dwBase];             // A 16-bit layout
                } else {
#pragma unroll
                    for (int v = 0; v < 8; ++v) a.u[v] = 0u;
                }

                int tap = (dz * 3 + dy) * 3 + dx;
                const unsigned int* bp = &sw[(tap * 32 + mrow) * 16];
                Frag16 b0, b1;
#pragma unroll
                for (int v = 0; v < 8; ++v) { b0.u[v] = bp[v + (v & 4) + dwBase]; b1.u[v] = bp[256 + v + (v & 4) + dwBase]; }

                acc0 = __builtin_amdgcn_wmma_f32_16x16x32_bf16(
                           false, a.bf, false, b0.bf, (short)0, acc0, false, false);
                acc1 = __builtin_amdgcn_wmma_f32_16x16x32_bf16(
                           false, a.bf, false, b1.bf, (short)0, acc1, false, false);
            }
        }
    }

    // Epilogue: C layout — VGPR r holds M = r + 8*half, lane%16 = N
    const int n0 = mrow;
    const int n1 = mrow + 16;
    float s0 = sb[n0],  b0v = sb[32 + n0];
    float s1 = sb[n1],  b1v = sb[32 + n1];
#pragma unroll
    for (int r = 0; r < 8; ++r) {
        int vC = voxBase + r + (half << 3);
        float mk = mko[vC];
        float f0 = fmaxf(acc0[r] * s0 + b0v, 0.0f) * mk;
        float f1 = fmaxf(acc1[r] * s1 + b1v, 0.0f) * mk;
        out[(size_t)vC * 32 + n0] = f2bf(f0);
        out[(size_t)vC * 32 + n1] = f2bf(f1);
    }
}

// ---------------------------------------------------------------------------
// Trilinear sampling: one thread per point, 8 corners x 32 bf16 channels.
// Output layout (B, N, C): out[n*32 + c] f32.
// ---------------------------------------------------------------------------
__global__ void sample_tri(const unsigned short* __restrict__ vol,
                           const float* __restrict__ coords,
                           float* __restrict__ out, int N, int Dv) {
    int n = blockIdx.x * blockDim.x + threadIdx.x;
    if (n >= N) return;
    float gx = coords[n * 3 + 0];
    float gy = coords[n * 3 + 1];
    float gz = coords[n * 3 + 2];
    float fx = (gx + 1.0f) * 0.5f * (float)(Dv - 1);
    float fy = (gy + 1.0f) * 0.5f * (float)(Dv - 1);
    float fz = (gz + 1.0f) * 0.5f * (float)(Dv - 1);
    float x0f = floorf(fx), y0f = floorf(fy), z0f = floorf(fz);
    int x0 = (int)x0f, y0 = (int)y0f, z0 = (int)z0f;
    float wx1 = fx - x0f, wx0 = 1.0f - wx1;
    float wy1 = fy - y0f, wy0 = 1.0f - wy1;
    float wz1 = fz - z0f, wz0 = 1.0f - wz1;

    float acc[32];
#pragma unroll
    for (int c = 0; c < 32; ++c) acc[c] = 0.0f;

#pragma unroll
    for (int corner = 0; corner < 8; ++corner) {
        int dx = corner & 1, dy = (corner >> 1) & 1, dz = (corner >> 2) & 1;
        int xi = x0 + dx, yi = y0 + dy, zi = z0 + dz;
        float w = (dz ? wz1 : wz0) * (dy ? wy1 : wy0) * (dx ? wx1 : wx0);
        bool valid = (xi >= 0) && (xi < Dv) && (yi >= 0) && (yi < Dv) &&
                     (zi >= 0) && (zi < Dv);
        float ww = valid ? w : 0.0f;
        int xc = min(max(xi, 0), Dv - 1);
        int yc = min(max(yi, 0), Dv - 1);
        int zc = min(max(zi, 0), Dv - 1);
        const unsigned int* row =
            (const unsigned int*)vol + (size_t)((zc * Dv + yc) * Dv + xc) * 16;
#pragma unroll
        for (int i = 0; i < 16; ++i) {
            unsigned int d = row[i];
            acc[2 * i]     += bf2f_lo(d) * ww;
            acc[2 * i + 1] += bf2f_hi(d) * ww;
        }
    }
#pragma unroll
    for (int c = 0; c < 32; ++c) out[(size_t)n * 32 + c] = acc[c];
}

// ---------------------------------------------------------------------------
// Host orchestration
// ---------------------------------------------------------------------------
extern "C" void kernel_launch(void* const* d_in, const int* in_sizes, int n_in,
                              void* d_out, int out_size, void* d_ws, size_t ws_size,
                              hipStream_t stream) {
    const float* x      = (const float*)d_in[0];
    const float* coords = (const float*)d_in[1];
    const float* w      = (const float*)d_in[2];
    const float* gamma  = (const float*)d_in[3];
    const float* beta   = (const float*)d_in[4];
    const float* mean   = (const float*)d_in[5];
    const float* var    = (const float*)d_in[6];
    const int*   mask   = (const int*)d_in[7];

    const int N  = in_sizes[1] / 3;     // 65536 sample points
    const int D0 = 96, D1 = 48;
    const int V0 = D0 * D0 * D0;        // 884736
    const int V1 = D1 * D1 * D1;        // 110592

    char* ws = (char*)d_ws;
    size_t off = 0;
    auto carve = [&](size_t bytes) -> void* {
        void* p = (void*)(ws + off);
        off += (bytes + 255) & ~(size_t)255;
        return p;
    };
    unsigned short* bufA = (unsigned short*)carve((size_t)V0 * 32 * 2);   // 56.6 MB
    unsigned short* bufB = (unsigned short*)carve((size_t)V0 * 32 * 2);   // 56.6 MB
    float*          m0f  = (float*)carve((size_t)V0 * 4);
    float*          m1f  = (float*)carve((size_t)V1 * 4);
    unsigned short* wbf  = (unsigned short*)carve((size_t)5 * 27 * 32 * 32 * 2);
    float*          sb   = (float*)carve((size_t)5 * 64 * 4);
    (void)ws_size; (void)n_in; (void)out_size;

    // prep
    x_pack <<<(32 * V0 + 255) / 256, 256, 0, stream>>>(x, mask, bufA, V0);
    m_pack <<<(V0 + 255) / 256,      256, 0, stream>>>(mask, m0f, V0);
    m_down <<<(V1 + 255) / 256,      256, 0, stream>>>(m0f, m1f, D0, D1);
    w_pack <<<(5 * 32 * 32 * 27 + 255) / 256, 256, 0, stream>>>(w, wbf);
    bn_pack<<<1, 256, 0, stream>>>(gamma, beta, mean, var, sb);

    const int WPL = 27 * 32 * 32;       // weights per layer (bf16 elems)
    // layer 0,1: 96^3 stride 1
    conv3_wmma<<<V0 / 128, 256, 0, stream>>>(bufA, bufB, wbf + 0 * WPL, sb + 0 * 64, m0f, D0, D0, 1);
    conv3_wmma<<<V0 / 128, 256, 0, stream>>>(bufB, bufA, wbf + 1 * WPL, sb + 1 * 64, m0f, D0, D0, 1);
    // layer 2: 96^3 -> 48^3 stride 2
    conv3_wmma<<<V1 / 128, 256, 0, stream>>>(bufA, bufB, wbf + 2 * WPL, sb + 2 * 64, m1f, D0, D1, 2);
    // layer 3,4: 48^3 stride 1
    conv3_wmma<<<V1 / 128, 256, 0, stream>>>(bufB, bufA, wbf + 3 * WPL, sb + 3 * 64, m1f, D1, D1, 1);
    conv3_wmma<<<V1 / 128, 256, 0, stream>>>(bufA, bufB, wbf + 4 * WPL, sb + 4 * 64, m1f, D1, D1, 1);

    // trilinear gather -> (N, 32) f32
    sample_tri<<<(N + 255) / 256, 256, 0, stream>>>(bufB, coords, (float*)d_out, N, D1);
}
